// MPNN_38843684225500
// MI455X (gfx1250) — compile-verified, matrix operand order checked by reading, not run on hardware
//
#include <hip/hip_runtime.h>
#include <hip/hip_bf16.h>

typedef __attribute__((ext_vector_type(16))) _Float16 v16h;
typedef __attribute__((ext_vector_type(8)))  float    v8f;

#define NN 10000
#define EE 20000
#define BB 400
#define DD 64

// ---------------------------------------------------------------- utilities

__device__ __forceinline__ float sigmf(float x) { return 1.0f / (1.0f + __expf(-x)); }

__device__ __forceinline__ void atomicMaxF(float* addr, float v) {
    int* ia = (int*)addr;
    int old = __float_as_int(*addr);
    while (__int_as_float(old) < v) {
        int assumed = old;
        old = atomicCAS(ia, assumed, __float_as_int(v));
        if (old == assumed) break;
    }
}

__global__ void zerof_kernel(float* p, int n) {
    int i = blockIdx.x * blockDim.x + threadIdx.x;
    if (i < n) p[i] = 0.0f;
}

// f32 [rows,K] -> f16 [rows,Kp] with zero pad (Kp >= K)
__global__ void cvt_pad_kernel(_Float16* __restrict__ dst, const float* __restrict__ src,
                               int rows, int K, int Kp) {
    int i = blockIdx.x * blockDim.x + threadIdx.x;
    int total = rows * Kp;
    if (i >= total) return;
    int r = i / Kp, k = i - r * Kp;
    dst[i] = (k < K) ? (_Float16)src[(size_t)r * K + k] : (_Float16)0.0f;
}

// ---------------------------------------------------------------- WMMA GEMM
// C[M,F] = act(A[M,K] @ W[F,K]^T + bias), A/W f16 row-major, K % 32 == 0,
// M % 16 == 0, F % (16*NT) == 0.  One wave computes a 16 x (16*NT) C block:
// one A-fragment load feeds NT independent WMMAs (A reuse, latency hiding).
// Fragment load (A and W identical pattern, ISA 16-bit A layout):
//   lane<16 : row = lane,    K = {0..7, 16..23} (+kb)
//   lane>=16: row = lane-16, K = {8..15,24..31} (+kb)

__device__ __forceinline__ v16h frag_load(const _Float16* __restrict__ p, int lane_hi) {
    union { float4 q[2]; v16h v; } u;
    const _Float16* pp = p + (lane_hi << 3);
    u.q[0] = *(const float4*)(pp);
    u.q[1] = *(const float4*)(pp + 16);
    return u.v;
}

template <int NT>
__global__ __launch_bounds__(32) void wmma_gemm_f16(
    const _Float16* __restrict__ A, const _Float16* __restrict__ W,
    const float* __restrict__ bias, float* __restrict__ Cf, _Float16* __restrict__ Ch,
    int K, int F, int do_relu) {
    const int lane = threadIdx.x & 31;
    const int n0 = blockIdx.x * (16 * NT);
    const int m0 = blockIdx.y << 4;
    const int lr = lane & 15;
    const int lh = lane >> 4;

    v8f acc[NT];
#pragma unroll
    for (int nt = 0; nt < NT; ++nt) acc[nt] = (v8f){};

    const _Float16* arow = A + (size_t)(m0 + lr) * K;
    const _Float16* wrow[NT];
#pragma unroll
    for (int nt = 0; nt < NT; ++nt)
        wrow[nt] = W + (size_t)(n0 + nt * 16 + lr) * K;

    for (int kb = 0; kb < K; kb += 32) {
        v16h a = frag_load(arow + kb, lh);
#pragma unroll
        for (int nt = 0; nt < NT; ++nt) {
            v16h b = frag_load(wrow[nt] + kb, lh);
            acc[nt] = __builtin_amdgcn_wmma_f32_16x16x32_f16(
                /*neg_a=*/false, a, /*neg_b=*/false, b,
                /*c_mod=*/(short)0, acc[nt], /*reuse_a=*/false, /*reuse_b=*/false);
        }
    }

#pragma unroll
    for (int nt = 0; nt < NT; ++nt) {
        const int col = n0 + nt * 16 + lr;
        const float bcol = bias[col];
#pragma unroll
        for (int r = 0; r < 8; ++r) {
            int row = m0 + r + (lh << 3);
            float v = acc[nt][r] + bcol;
            if (do_relu) v = v > 0.0f ? v : 0.0f;
            size_t off = (size_t)row * F + col;
            if (Cf) Cf[off] = v;
            if (Ch) Ch[off] = (_Float16)v;
        }
    }
}

// ---------------------------------------------------------------- graph conv

__global__ void count_kernel(const int* __restrict__ dst, float* __restrict__ cnt, int E) {
    int e = blockIdx.x * blockDim.x + threadIdx.x;
    if (e < E) atomicAdd(&cnt[dst[e]], 1.0f);
}

// per-edge GEMV msg[e] = out16[src[e]] (1x64) x ew[e] (64x64), scatter-add to agg[dst].
// One wave32 per edge: lane (dg = t>>3, fg = t&7) owns rows d = dg+4k and
// 8 contiguous columns [8*fg, 8*fg+8); each b128 load moves 512B contiguous
// per wave.  Cross-lane reduce over the 4 row-groups via shfl_xor(8|16).
__global__ __launch_bounds__(32) void msg_scatter_kernel(
    const _Float16* __restrict__ out16, const _Float16* __restrict__ ew,
    const int* __restrict__ src, const int* __restrict__ dst,
    float* __restrict__ agg) {
    const int e = blockIdx.x;
    const int t = threadIdx.x;
    const int fg = t & 7;
    const int dg = t >> 3;
    __shared__ float s[DD];
    {
        const _Float16* ov = out16 + (size_t)src[e] * DD;
        s[2 * t]     = (float)ov[2 * t];
        s[2 * t + 1] = (float)ov[2 * t + 1];
    }
    __syncthreads();

    const _Float16* base = ew + (size_t)e * (DD * DD) + fg * 8;
    float acc[8] = {};
#pragma unroll 4
    for (int d = dg; d < DD; d += 4) {
        union { float4 q; _Float16 h[8]; } u;
        u.q = *(const float4*)(base + (size_t)d * DD);
        const float sv = s[d];
#pragma unroll
        for (int i = 0; i < 8; ++i) acc[i] += sv * (float)u.h[i];
    }
#pragma unroll
    for (int i = 0; i < 8; ++i) {
        acc[i] += __shfl_xor(acc[i], 8, 32);
        acc[i] += __shfl_xor(acc[i], 16, 32);
    }
    if (t < 8) {
        float* ap = agg + (size_t)dst[e] * DD + fg * 8;
#pragma unroll
        for (int i = 0; i < 8; ++i) atomicAdd(&ap[i], acc[i]);
    }
}

// m16 = relu(agg / max(cnt,1) + b_conv)
__global__ __launch_bounds__(64) void agg_finalize_kernel(
    const float* __restrict__ agg, const float* __restrict__ cnt,
    const float* __restrict__ b_conv, _Float16* __restrict__ m16) {
    const int n = blockIdx.x, t = threadIdx.x;
    float c = cnt[n]; c = c < 1.0f ? 1.0f : c;
    float v = agg[(size_t)n * DD + t] / c + b_conv[t];
    v = v > 0.0f ? v : 0.0f;
    m16[(size_t)n * DD + t] = (_Float16)v;
}

// torch GRU cell, gate order r,z,n
__global__ __launch_bounds__(64) void gru_cell_kernel(
    const float* __restrict__ gi, const float* __restrict__ gh,
    float* __restrict__ h, _Float16* __restrict__ h16) {
    const int n = blockIdx.x, t = threadIdx.x;
    const float* gin = gi + (size_t)n * (3 * DD);
    const float* ghn = gh + (size_t)n * (3 * DD);
    float r = sigmf(gin[t] + ghn[t]);
    float z = sigmf(gin[DD + t] + ghn[DD + t]);
    float nn = tanhf(gin[2 * DD + t] + r * ghn[2 * DD + t]);
    size_t off = (size_t)n * DD + t;
    float hv = (1.0f - z) * nn + z * h[off];
    h[off] = hv;
    h16[off] = (_Float16)hv;
}

// ---------------------------------------------------------------- Set2Set

__global__ __launch_bounds__(256) void s2s_lstm_kernel(
    const float* __restrict__ qstar, float* __restrict__ hl, float* __restrict__ cl,
    const float* __restrict__ Wih, const float* __restrict__ bih,
    const float* __restrict__ Whh, const float* __restrict__ bhh) {
    const int b = blockIdx.x, j = threadIdx.x;
    __shared__ float g[4 * DD];
    const float* q = qstar + (size_t)b * (2 * DD);
    const float* h = hl + (size_t)b * DD;
    float acc = bih[j] + bhh[j];
    const float* wi = Wih + (size_t)j * (2 * DD);
    for (int k = 0; k < 2 * DD; ++k) acc += q[k] * wi[k];
    const float* wh = Whh + (size_t)j * DD;
    for (int k = 0; k < DD; ++k) acc += h[k] * wh[k];
    g[j] = acc;
    __syncthreads();
    if (j < DD) {
        float ig = sigmf(g[j]);
        float fg = sigmf(g[DD + j]);
        float gg = tanhf(g[2 * DD + j]);
        float og = sigmf(g[3 * DD + j]);
        size_t off = (size_t)b * DD + j;
        float c = fg * cl[off] + ig * gg;
        cl[off] = c;
        hl[off] = og * tanhf(c);
    }
}

__global__ __launch_bounds__(64) void s2s_init_kernel(
    float* __restrict__ mmax, float* __restrict__ denom, float* __restrict__ rbuf) {
    const int b = blockIdx.x, t = threadIdx.x;
    rbuf[(size_t)b * DD + t] = 0.0f;
    if (t == 0) { mmax[b] = -3.4e38f; denom[b] = 0.0f; }
}

__global__ void attn_e_kernel(const float* __restrict__ out, const float* __restrict__ hl,
                              const int* __restrict__ batch, float* __restrict__ e,
                              float* __restrict__ mmax, int N) {
    int n = blockIdx.x * blockDim.x + threadIdx.x;
    if (n >= N) return;
    int b = batch[n];
    const float* o = out + (size_t)n * DD;
    const float* q = hl + (size_t)b * DD;
    float s = 0.0f;
#pragma unroll 8
    for (int k = 0; k < DD; ++k) s += o[k] * q[k];
    e[n] = s;
    atomicMaxF(&mmax[b], s);
}

__global__ void attn_exp_kernel(float* __restrict__ e, const float* __restrict__ mmax,
                                const int* __restrict__ batch, float* __restrict__ denom, int N) {
    int n = blockIdx.x * blockDim.x + threadIdx.x;
    if (n >= N) return;
    int b = batch[n];
    float a = __expf(e[n] - mmax[b]);
    e[n] = a;
    atomicAdd(&denom[b], a);
}

__global__ __launch_bounds__(64) void attn_r_kernel(
    const float* __restrict__ e, const float* __restrict__ out,
    const int* __restrict__ batch, float* __restrict__ rbuf) {
    const int n = blockIdx.x, t = threadIdx.x;
    int b = batch[n];
    atomicAdd(&rbuf[(size_t)b * DD + t], e[n] * out[(size_t)n * DD + t]);
}

__global__ __launch_bounds__(64) void qstar_kernel(
    float* __restrict__ qstar, const float* __restrict__ hl,
    const float* __restrict__ rbuf, const float* __restrict__ denom) {
    const int b = blockIdx.x, t = threadIdx.x;
    qstar[(size_t)b * (2 * DD) + t] = hl[(size_t)b * DD + t];
    qstar[(size_t)b * (2 * DD) + DD + t] = rbuf[(size_t)b * DD + t] / denom[b];
}

// ---------------------------------------------------------------- output head

__global__ __launch_bounds__(64) void head_kernel(
    const float* __restrict__ qstar, const float* __restrict__ Wo1,
    const float* __restrict__ bo1, const float* __restrict__ Wo2,
    const float* __restrict__ bo2, float* __restrict__ out) {
    const int b = blockIdx.x, j = threadIdx.x;
    __shared__ float hsh[DD];
    const float* q = qstar + (size_t)b * (2 * DD);
    const float* w = Wo1 + (size_t)j * (2 * DD);
    float acc = bo1[j];
    for (int k = 0; k < 2 * DD; ++k) acc += q[k] * w[k];
    acc = acc > 0.0f ? acc : 0.0f;
    hsh[j] = acc * Wo2[j];
    __syncthreads();
    if (j == 0) {
        float s = 0.0f;
        for (int k = 0; k < DD; ++k) s += hsh[k];
        out[b] = s + bo2[0];
    }
}

// ---------------------------------------------------------------- launch

extern "C" void kernel_launch(void* const* d_in, const int* in_sizes, int n_in,
                              void* d_out, int out_size, void* d_ws, size_t ws_size,
                              hipStream_t stream) {
    const float* x        = (const float*)d_in[0];
    const float* ea       = (const float*)d_in[1];
    const float* W_in     = (const float*)d_in[2];
    const float* b_in     = (const float*)d_in[3];
    const float* W_e1     = (const float*)d_in[4];
    const float* b_e1     = (const float*)d_in[5];
    const float* W_e2     = (const float*)d_in[6];
    const float* b_e2     = (const float*)d_in[7];
    const float* b_conv   = (const float*)d_in[8];
    const float* W_ih     = (const float*)d_in[9];
    const float* b_ih     = (const float*)d_in[10];
    const float* W_hh     = (const float*)d_in[11];
    const float* b_hh     = (const float*)d_in[12];
    const float* W_ih_l   = (const float*)d_in[13];
    const float* b_ih_l   = (const float*)d_in[14];
    const float* W_hh_l   = (const float*)d_in[15];
    const float* b_hh_l   = (const float*)d_in[16];
    const float* W_o1     = (const float*)d_in[17];
    const float* b_o1     = (const float*)d_in[18];
    const float* W_o2     = (const float*)d_in[19];
    const float* b_o2     = (const float*)d_in[20];
    const int*   eidx     = (const int*)d_in[21];
    const int*   batch    = (const int*)d_in[22];
    float* out = (float*)d_out;

    const int* src = eidx;
    const int* dst = eidx + EE;

    // ---- workspace bump allocator (256B aligned)
    char* wsp = (char*)d_ws;
    auto alloc = [&](size_t bytes) -> void* {
        void* p = (void*)wsp;
        wsp += (bytes + 255) & ~(size_t)255;
        return p;
    };
    _Float16* x16   = (_Float16*)alloc((size_t)NN * 32 * 2);
    _Float16* ea16  = (_Float16*)alloc((size_t)EE * 32 * 2);
    _Float16* Win16 = (_Float16*)alloc((size_t)64 * 32 * 2);
    _Float16* We116 = (_Float16*)alloc((size_t)64 * 32 * 2);
    _Float16* We216 = (_Float16*)alloc((size_t)4096 * 64 * 2);
    _Float16* Wih16 = (_Float16*)alloc((size_t)192 * 64 * 2);
    _Float16* Whh16 = (_Float16*)alloc((size_t)192 * 64 * 2);
    _Float16* h116  = (_Float16*)alloc((size_t)EE * 64 * 2);
    _Float16* ew16  = (_Float16*)alloc((size_t)EE * 4096 * 2);   // 164 MB
    _Float16* out16 = (_Float16*)alloc((size_t)NN * 64 * 2);
    _Float16* m16   = (_Float16*)alloc((size_t)NN * 64 * 2);
    float* hbuf  = (float*)alloc((size_t)NN * 64 * 4);
    float* gi    = (float*)alloc((size_t)NN * 192 * 4);
    float* gh    = (float*)alloc((size_t)NN * 192 * 4);
    float* agg   = (float*)alloc((size_t)NN * 64 * 4);
    float* cnt   = (float*)alloc((size_t)NN * 4);
    float* ebuf  = (float*)alloc((size_t)NN * 4);
    float* qstar = (float*)alloc((size_t)BB * 128 * 4);
    float* hl    = (float*)alloc((size_t)BB * 64 * 4);
    float* cl    = (float*)alloc((size_t)BB * 64 * 4);
    float* mmax  = (float*)alloc((size_t)BB * 4);
    float* denom = (float*)alloc((size_t)BB * 4);
    float* rbuf  = (float*)alloc((size_t)BB * 64 * 4);

    auto cvt = [&](_Float16* dstp, const float* srcp, int rows, int K, int Kp) {
        int total = rows * Kp;
        cvt_pad_kernel<<<(total + 255) / 256, 256, 0, stream>>>(dstp, srcp, rows, K, Kp);
    };

    // ---- f32 -> f16 conversions (zero-pad K=16 paths to 32)
    cvt(x16, x, NN, 32, 32);
    cvt(ea16, ea, EE, 16, 32);
    cvt(Win16, W_in, 64, 32, 32);
    cvt(We116, W_e1, 64, 16, 32);
    cvt(We216, W_e2, 4096, 64, 64);
    cvt(Wih16, W_ih, 192, 64, 64);
    cvt(Whh16, W_hh, 192, 64, 64);

    // ---- edge MLP: h1 = relu(ea @ We1^T + b_e1)   [E,64]
    wmma_gemm_f16<4><<<dim3(64 / 64, EE / 16), 32, 0, stream>>>(
        ea16, We116, b_e1, nullptr, h116, 32, 64, 1);
    // ---- ew = h1 @ We2^T + b_e2   [E,4096] f16   (the big GEMM, 10.5 GFLOP)
    wmma_gemm_f16<4><<<dim3(4096 / 64, EE / 16), 32, 0, stream>>>(
        h116, We216, b_e2, nullptr, ew16, 64, 4096, 0);
    // ---- out0 = relu(x @ W_in^T + b_in); h = out0
    wmma_gemm_f16<4><<<dim3(64 / 64, NN / 16), 32, 0, stream>>>(
        x16, Win16, b_in, hbuf, out16, 32, 64, 1);

    // ---- in-degree counts
    zerof_kernel<<<(NN + 255) / 256, 256, 0, stream>>>(cnt, NN);
    count_kernel<<<(EE + 255) / 256, 256, 0, stream>>>(dst, cnt, EE);

    // ---- 3x (NNConv + GRU)
    for (int it = 0; it < 3; ++it) {
        zerof_kernel<<<(NN * 64 + 255) / 256, 256, 0, stream>>>(agg, NN * 64);
        msg_scatter_kernel<<<EE, 32, 0, stream>>>(out16, ew16, src, dst, agg);
        agg_finalize_kernel<<<NN, 64, 0, stream>>>(agg, cnt, b_conv, m16);
        wmma_gemm_f16<4><<<dim3(192 / 64, NN / 16), 32, 0, stream>>>(
            m16, Wih16, b_ih, gi, nullptr, 64, 192, 0);
        wmma_gemm_f16<4><<<dim3(192 / 64, NN / 16), 32, 0, stream>>>(
            out16, Whh16, b_hh, gh, nullptr, 64, 192, 0);
        gru_cell_kernel<<<NN, 64, 0, stream>>>(gi, gh, hbuf, out16);
    }

    // ---- Set2Set (3 steps)
    zerof_kernel<<<(BB * 128 + 255) / 256, 256, 0, stream>>>(qstar, BB * 128);
    zerof_kernel<<<(BB * 64 + 255) / 256, 256, 0, stream>>>(hl, BB * 64);
    zerof_kernel<<<(BB * 64 + 255) / 256, 256, 0, stream>>>(cl, BB * 64);
    for (int s = 0; s < 3; ++s) {
        s2s_lstm_kernel<<<BB, 256, 0, stream>>>(qstar, hl, cl, W_ih_l, b_ih_l, W_hh_l, b_hh_l);
        s2s_init_kernel<<<BB, 64, 0, stream>>>(mmax, denom, rbuf);
        attn_e_kernel<<<(NN + 127) / 128, 128, 0, stream>>>(hbuf, hl, batch, ebuf, mmax, NN);
        attn_exp_kernel<<<(NN + 127) / 128, 128, 0, stream>>>(ebuf, mmax, batch, denom, NN);
        attn_r_kernel<<<NN, 64, 0, stream>>>(ebuf, hbuf, batch, rbuf);
        qstar_kernel<<<BB, 64, 0, stream>>>(qstar, hl, rbuf, denom);
    }

    // ---- output head
    head_kernel<<<BB, 64, 0, stream>>>(qstar, W_o1, b_o1, W_o2, b_o2, out);
}
